// GraphAffineCoupling_26001732010203
// MI455X (gfx1250) — matrix operand order repaired and checked by reading, not compiled, not run on hardware
//
#include <hip/hip_runtime.h>
#include <math.h>

// ---- problem constants (ZINC-250k config) ----
#define B_BATCH 4096
#define N_NODE  38
#define CIN     9
#define E_EDGE  4
#define H0      256
#define H1      512
#define H2      64
#define M_ROWS  (B_BATCH * N_NODE)   // 155648, divisible by 16
#define BN_EPS  1e-5f

typedef __attribute__((ext_vector_type(16))) __bf16 v16bf;
typedef __attribute__((ext_vector_type(8)))  float  v8f;

union Frag { v16bf v; uint4 q[2]; };

__device__ __forceinline__ unsigned short f2bf(float f) {
  unsigned u = __float_as_uint(f);
  u += 0x7FFFu + ((u >> 16) & 1u);        // round-to-nearest-even
  return (unsigned short)(u >> 16);
}
__device__ __forceinline__ float bf2f(unsigned short h) {
  return __uint_as_float(((unsigned)h) << 16);
}

// ---------------------------------------------------------------------------
// zero the BN-stat accumulators (re-run every call: deterministic)
// ---------------------------------------------------------------------------
__global__ void k_zero(float* p, int n) {
  int i = blockIdx.x * blockDim.x + threadIdx.x;
  if (i < n) p[i] = 0.f;
}

// ---------------------------------------------------------------------------
// Pack fused GraphConv weight W0 (K=64 x 256) into WMMA B-fragment order.
// rows 0..8   : w_self[k][c]
// rows 9..44  : w_edge[k][c*4+e]  (row = 9 + e*9 + k)   -- einsum factorization
// rows 45..48 : b_edge[c*4+e]     (paired with adj row-sums in A)
// rows 49..63 : 0
// Fragment order: [kstep][tile][lane][j], elem j of lane L is K = ks*32+(L>>4)*16+j,
// col = tile*16 + (L&15)  (contiguous k order, same as the A loads in k_gemm).
// ---------------------------------------------------------------------------
__global__ void k_pack_w0(const float* __restrict__ w_self,
                          const float* __restrict__ w_edge,
                          const float* __restrict__ b_edge,
                          unsigned short* __restrict__ W0f) {
  int idx = blockIdx.x * blockDim.x + threadIdx.x;          // 2*16*32*16 = 16384
  if (idx >= 2 * 16 * 32 * 16) return;
  int j    = idx & 15;
  int lane = (idx >> 4) & 31;
  int rest = idx >> 9;
  int tile = rest & 15;
  int ks   = rest >> 4;
  int col  = tile * 16 + (lane & 15);
  int kk   = ks * 32 + (lane >> 4) * 16 + j;
  float v = 0.f;
  if (kk < 9) {
    v = w_self[kk * H0 + col];
  } else if (kk < 45) {
    int ek = kk - 9, e = ek / 9, k = ek % 9;
    v = w_edge[k * (H0 * E_EDGE) + col * E_EDGE + e];
  } else if (kk < 49) {
    int e = kk - 45;
    v = b_edge[col * E_EDGE + e];
  }
  W0f[idx] = f2bf(v);
}

// generic dense weight (K x Ncols f32, K multiple of 32, Ncols multiple of 16)
__global__ void k_pack_dense(const float* __restrict__ W,
                             unsigned short* __restrict__ Bf,
                             int K, int Ncols) {
  int total = K * Ncols;
  int ntiles = Ncols >> 4;
  for (int idx = blockIdx.x * blockDim.x + threadIdx.x; idx < total;
       idx += gridDim.x * blockDim.x) {
    int j    = idx & 15;
    int lane = (idx >> 4) & 31;
    int rest = idx >> 9;
    int tile = rest % ntiles;
    int ks   = rest / ntiles;
    int col  = tile * 16 + (lane & 15);
    int kk   = ks * 32 + (lane >> 4) * 16 + j;
    Bf[idx] = f2bf(W[kk * Ncols + col]);
  }
}

// ---------------------------------------------------------------------------
// Stage 0: build bf16 A matrix (M_ROWS x 64) for the fused GraphConv GEMM.
// cols 0..8   : masked_x[b,i,k]
// cols 9..44  : P[b,i,e*9+k] = sum_j adj[b,e,i,j] * masked_x[b,j,k]
// cols 45..48 : rowsum_e = sum_j adj[b,e,i,j]     (carries b_edge)
// cols 49..63 : 0
// One block per batch; adj[b] (23 KB) staged in LDS.
// ---------------------------------------------------------------------------
__global__ __launch_bounds__(256)
void k_stage0(const float* __restrict__ adj, const float* __restrict__ x,
              const float* __restrict__ mask, unsigned short* __restrict__ Abuf) {
  __shared__ float s_adj[E_EDGE * N_NODE * N_NODE];   // 5776 floats
  __shared__ float s_mx[N_NODE * CIN];                // 342 floats
  const int tid = threadIdx.x;
  const int b   = blockIdx.x;
  const float* adjb = adj + (size_t)b * (E_EDGE * N_NODE * N_NODE);
  for (int i = tid; i < E_EDGE * N_NODE * N_NODE; i += 256) s_adj[i] = adjb[i];
  for (int i = tid; i < N_NODE * CIN; i += 256)
    s_mx[i] = x[(size_t)b * (N_NODE * CIN) + i] * mask[i];
  __syncthreads();

  for (int idx = tid; idx < N_NODE * 64; idx += 256) {
    int i = idx >> 6;
    int c = idx & 63;
    float v = 0.f;
    if (c < 9) {
      v = s_mx[i * CIN + c];
    } else if (c < 45) {
      int ek = c - 9, e = ek / 9, k = ek % 9;
      const float* ar = &s_adj[(e * N_NODE + i) * N_NODE];
      float acc = 0.f;
#pragma unroll
      for (int j = 0; j < N_NODE; ++j) acc += ar[j] * s_mx[j * CIN + k];
      v = acc;
    } else if (c < 49) {
      int e = c - 45;
      const float* ar = &s_adj[(e * N_NODE + i) * N_NODE];
      float acc = 0.f;
#pragma unroll
      for (int j = 0; j < N_NODE; ++j) acc += ar[j];
      v = acc;
    }
    Abuf[((size_t)b * N_NODE + i) * 64 + c] = f2bf(v);
  }
}

// ---------------------------------------------------------------------------
// bf16 WMMA GEMM: C[M x NC] = A[M x K] * Bfrag + bias, C stored as bf16.
// Block = nwaves*32 threads, one 16-row tile per block; each wave owns TPW
// 16-col tiles. B pre-packed in fragment order -> only contiguous b128 loads.
// ---------------------------------------------------------------------------
template <int TPW, int KSTEPS, int NTILES>
__global__ __launch_bounds__(256)
void k_gemm(const unsigned short* __restrict__ A,
            const unsigned short* __restrict__ Bf,
            const float* __restrict__ bias,
            unsigned short* __restrict__ Cp) {
  constexpr int K  = KSTEPS * 32;
  constexpr int NC = NTILES * 16;
  const int lane = threadIdx.x & 31;
  const int wave = threadIdx.x >> 5;
  const int m  = lane & 15;
  const int kh = lane >> 4;           // 0: K 0-15, 1: K 16-31
  const size_t rowTile = blockIdx.x;

  v8f acc[TPW] = {};
  const unsigned short* arow = A + (rowTile * 16 + m) * (size_t)K + kh * 16;

#pragma unroll
  for (int ks = 0; ks < KSTEPS; ++ks) {
    Frag a;
    a.q[0] = *(const uint4*)(arow + ks * 32);
    a.q[1] = *(const uint4*)(arow + ks * 32 + 8);
#pragma unroll
    for (int t = 0; t < TPW; ++t) {
      const int tile = wave * TPW + t;
      const unsigned short* bp =
          Bf + (((size_t)ks * NTILES + tile) * 32 + lane) * 16;
      Frag bfr;
      bfr.q[0] = *(const uint4*)(bp);
      bfr.q[1] = *(const uint4*)(bp + 8);
      acc[t] = __builtin_amdgcn_wmma_f32_16x16x32_bf16(
          false, a.v, false, bfr.v, (short)0, acc[t], false, false);
    }
  }

#pragma unroll
  for (int t = 0; t < TPW; ++t) {
    const int tile = wave * TPW + t;
    const int col  = tile * 16 + m;
    const float bv = bias[col];
    const size_t rowbase = rowTile * 16 + kh * 8;   // C/D: VGPR r -> M = kh*8 + r
#pragma unroll
    for (int r = 0; r < 8; ++r) {
      float v = acc[t][r] + bv;
      Cp[(rowbase + r) * NC + col] = f2bf(v);
    }
  }
}

// ---------------------------------------------------------------------------
// BatchNorm stats per node over (batch, channels): sum & sumsq -> atomics
// ---------------------------------------------------------------------------
__global__ __launch_bounds__(256)
void k_bnstats(const unsigned short* __restrict__ pre, float* __restrict__ stat,
               int C) {
  __shared__ float rs[256];
  __shared__ float rq[256];
  const int n = blockIdx.x;
  float s = 0.f, q = 0.f;
  for (int b = blockIdx.y; b < B_BATCH; b += gridDim.y) {
    const unsigned short* row = pre + (size_t)(b * N_NODE + n) * C;
    for (int c = threadIdx.x; c < C; c += blockDim.x) {
      float v = bf2f(row[c]);
      s += v; q += v * v;
    }
  }
  rs[threadIdx.x] = s; rq[threadIdx.x] = q;
  __syncthreads();
  for (int off = 128; off > 0; off >>= 1) {
    if ((int)threadIdx.x < off) {
      rs[threadIdx.x] += rs[threadIdx.x + off];
      rq[threadIdx.x] += rq[threadIdx.x + off];
    }
    __syncthreads();
  }
  if (threadIdx.x == 0) {
    atomicAdd(&stat[n], rs[0]);
    atomicAdd(&stat[N_NODE + n], rq[0]);
  }
}

// apply BN (biased var) + ReLU, cast to bf16
__global__ __launch_bounds__(256)
void k_bnapply(const unsigned short* __restrict__ pre,
               unsigned short* __restrict__ post,
               const float* __restrict__ stat,
               const float* __restrict__ w, const float* __restrict__ bb,
               int shiftC, float invCnt, unsigned total) {
  for (unsigned idx = blockIdx.x * blockDim.x + threadIdx.x; idx < total;
       idx += gridDim.x * blockDim.x) {
    unsigned row = idx >> shiftC;
    int n = row % N_NODE;
    float mean = stat[n] * invCnt;
    float var  = stat[N_NODE + n] * invCnt - mean * mean;
    float v = (bf2f(pre[idx]) - mean) * rsqrtf(var + BN_EPS) * w[n] + bb[n];
    post[idx] = f2bf(v > 0.f ? v : 0.f);
  }
}

// ---------------------------------------------------------------------------
// Final: lin_out (64 -> 18), sigmoid, masked affine coupling, logdet per batch
// ---------------------------------------------------------------------------
__global__ __launch_bounds__(256)
void k_final(const unsigned short* __restrict__ h2n,
             const float* __restrict__ lw, const float* __restrict__ lb,
             const float* __restrict__ x, const float* __restrict__ mask,
             float* __restrict__ out, float* __restrict__ logdet) {
  __shared__ float s_h2[N_NODE * H2];        // 2432
  __shared__ float s_lw[H2 * 2 * CIN];       // 1152
  __shared__ float s_lb[2 * CIN];
  __shared__ float s_x[N_NODE * CIN];
  __shared__ float s_mk[N_NODE * CIN];
  __shared__ float s_red[256];
  const int tid = threadIdx.x;
  const int b = blockIdx.x;

  for (int i = tid; i < N_NODE * H2; i += 256)
    s_h2[i] = bf2f(h2n[(size_t)b * (N_NODE * H2) + i]);
  for (int i = tid; i < H2 * 2 * CIN; i += 256) s_lw[i] = lw[i];
  if (tid < 2 * CIN) s_lb[tid] = lb[tid];
  for (int i = tid; i < N_NODE * CIN; i += 256) {
    s_x[i]  = x[(size_t)b * (N_NODE * CIN) + i];
    s_mk[i] = mask[i];
  }
  __syncthreads();

  float ld = 0.f;
  for (int idx = tid; idx < N_NODE * CIN; idx += 256) {
    int i = idx / CIN, c = idx % CIN;
    float a  = s_lb[c];
    float tt = s_lb[CIN + c];
    const float* hrow = &s_h2[i * H2];
#pragma unroll 8
    for (int k = 0; k < H2; ++k) {
      float hv = hrow[k];
      a  += hv * s_lw[k * (2 * CIN) + c];
      tt += hv * s_lw[k * (2 * CIN) + CIN + c];
    }
    float s  = 1.f / (1.f + expf(-a));
    float mk = s_mk[idx], xv = s_x[idx];
    out[(size_t)b * (N_NODE * CIN) + idx] = mk * xv + (1.f - mk) * (xv + tt) * s;
    ld += logf(fabsf(s));
  }
  s_red[tid] = ld;
  __syncthreads();
  for (int off = 128; off > 0; off >>= 1) {
    if (tid < off) s_red[tid] += s_red[tid + off];
    __syncthreads();
  }
  if (tid == 0) logdet[b] = s_red[0];
}

// ---------------------------------------------------------------------------
extern "C" void kernel_launch(void* const* d_in, const int* in_sizes, int n_in,
                              void* d_out, int out_size, void* d_ws, size_t ws_size,
                              hipStream_t stream) {
  (void)in_sizes; (void)n_in; (void)out_size; (void)ws_size;
  const float* adj      = (const float*)d_in[0];
  const float* x        = (const float*)d_in[1];
  const float* w_self   = (const float*)d_in[2];
  const float* b_self   = (const float*)d_in[3];
  const float* w_edge   = (const float*)d_in[4];
  const float* b_edge   = (const float*)d_in[5];
  const float* bn0_w    = (const float*)d_in[6];
  const float* bn0_b    = (const float*)d_in[7];
  const float* lin1_w   = (const float*)d_in[8];
  const float* lin1_b   = (const float*)d_in[9];
  const float* bn1_w    = (const float*)d_in[10];
  const float* bn1_b    = (const float*)d_in[11];
  const float* lin2_w   = (const float*)d_in[12];
  const float* lin2_b   = (const float*)d_in[13];
  const float* bn2_w    = (const float*)d_in[14];
  const float* bn2_b    = (const float*)d_in[15];
  const float* lout_w   = (const float*)d_in[16];
  const float* lout_b   = (const float*)d_in[17];
  const float* mask     = (const float*)d_in[18];

  char* ws = (char*)d_ws;
  unsigned short* W0f  = (unsigned short*)(ws);                         // 32 KB
  unsigned short* W1f  = (unsigned short*)(ws + (32 << 10));            // 256 KB
  unsigned short* W2f  = (unsigned short*)(ws + (288 << 10));           // 64 KB
  float*          stat = (float*)(ws + (352 << 10));                    // 3*128 f32
  float *st0 = stat, *st1 = stat + 128, *st2 = stat + 256;
  unsigned short* Abuf = (unsigned short*)(ws + ((size_t)1 << 20));     // 19.9 MB
  unsigned short* PRE  = (unsigned short*)(ws + ((size_t)32 << 20));    // 159.4 MB
  unsigned short* POST = (unsigned short*)(ws + ((size_t)196 << 20));   // 159.4 MB

  float* outp   = (float*)d_out;
  float* logdet = outp + (size_t)B_BATCH * N_NODE * CIN;

  // weight packing + stat zeroing
  k_zero<<<2, 256, 0, stream>>>(stat, 384);
  k_pack_w0<<<64, 256, 0, stream>>>(w_self, w_edge, b_edge, W0f);
  k_pack_dense<<<512, 256, 0, stream>>>(lin1_w, W1f, H0, H1);
  k_pack_dense<<<128, 256, 0, stream>>>(lin2_w, W2f, H1, H2);

  // GraphConv A-matrix (adj factorization)
  k_stage0<<<B_BATCH, 256, 0, stream>>>(adj, x, mask, Abuf);

  // stage 0: fused GraphConv GEMM (K=64 -> 256) + BN0 + ReLU
  k_gemm<2, 2, 16><<<M_ROWS / 16, 256, 0, stream>>>(Abuf, W0f, b_self, PRE);
  k_bnstats<<<dim3(N_NODE, 64), 256, 0, stream>>>(PRE, st0, H0);
  k_bnapply<<<4096, 256, 0, stream>>>(PRE, POST, st0, bn0_w, bn0_b, 8,
                                      1.f / (float)(B_BATCH * H0),
                                      (unsigned)M_ROWS * (unsigned)H0);

  // stage 1: 256 -> 512 + BN1 + ReLU
  k_gemm<4, 8, 32><<<M_ROWS / 16, 256, 0, stream>>>(POST, W1f, lin1_b, PRE);
  k_bnstats<<<dim3(N_NODE, 64), 256, 0, stream>>>(PRE, st1, H1);
  k_bnapply<<<4096, 256, 0, stream>>>(PRE, POST, st1, bn1_w, bn1_b, 9,
                                      1.f / (float)(B_BATCH * H1),
                                      (unsigned)M_ROWS * (unsigned)H1);

  // stage 2: 512 -> 64 + BN2 + ReLU
  k_gemm<1, 16, 4><<<M_ROWS / 16, 128, 0, stream>>>(POST, W2f, lin2_b, PRE);
  k_bnstats<<<dim3(N_NODE, 64), 256, 0, stream>>>(PRE, st2, H2);
  k_bnapply<<<4096, 256, 0, stream>>>(PRE, POST, st2, bn2_w, bn2_b, 6,
                                      1.f / (float)(B_BATCH * H2),
                                      (unsigned)M_ROWS * (unsigned)H2);

  // final affine coupling + logdet
  k_final<<<B_BATCH, 256, 0, stream>>>(POST, lout_w, lout_b, x, mask, outp, logdet);
}
